// Model_CFC_15616501088843
// MI455X (gfx1250) — matrix operand order repaired and checked by reading, not compiled, not run on hardware
//
// MI455X / gfx1250 end-to-end implementation of Model_CFC.
// GEMMs use v_wmma_f32_16x16x32_bf16 (bf16 inputs, f32 accumulate): CDNA5's
// f32 WMMA is only 16x16x4, so bf16 at K=32/instruction is the matrix-pipe
// path the chip rewards. Edge-MLP GEMMs (103 of ~140 GFLOP) are chunked in
// 32768-edge tiles so intermediates live in the 192MB L2.
#include <hip/hip_runtime.h>
#include <math.h>

typedef __attribute__((ext_vector_type(16))) __bf16 v16bf;
typedef __attribute__((ext_vector_type(8)))  float  v8f;

union FragAB { unsigned u[8]; v16bf v; };

__device__ __forceinline__ unsigned f2bf_pack(float lo, float hi) {
  unsigned ul = __float_as_uint(lo), uh = __float_as_uint(hi);
  ul = (ul + 0x7FFFu + ((ul >> 16) & 1u)) >> 16;   // RNE f32->bf16
  uh = (uh + 0x7FFFu + ((uh >> 16) & 1u)) >> 16;
  return ul | (uh << 16);
}

__device__ __forceinline__ float sspf(float x) {      // shifted softplus
  float sp = (x > 20.f) ? x : log1pf(__expf(x));
  return sp - 0.69314718055994531f;
}

template <int EPI> __device__ __forceinline__ float epilogue(float x) {
  if (EPI == 1) return fmaxf(x, 0.f);     // relu
  if (EPI == 2) return sspf(x);           // ssp
  if (EPI == 3) return tanhf(sspf(x));    // conv out
  if (EPI == 4) return tanhf(x);          // mlp
  return x;
}

// ---------------------------------------------------------------------------
// C[M,N] = epi(A[M,K] * ascale @ B[K,N] + bias). Block tile 128x64, 8 waves,
// wave w owns rows [16w,16w+16) x 64 cols as 4 accumulators (A frag reused
// against 4 B frags). Fragment layouts follow CDNA5 ISA 7.12.2 (wave32).
// LDS tiles hold packed bf16 pairs; K loop split into unguarded full tiles
// plus one branch-free (clamp+select) tail tile.
// ---------------------------------------------------------------------------
template <int EPI>
__global__ __launch_bounds__(256)
void gemm_wmma_bf16(const float* __restrict__ A, const float* __restrict__ Bw,
                    const float* __restrict__ bias, float* __restrict__ C,
                    int M, int N, int K, float ascale)
{
  __shared__ unsigned Asu[128][17];   // [m][kpair], +1 pad word (17 mod 64 banks)
  __shared__ unsigned Bsu[64][17];    // [n][kpair] (transposed: pairs contiguous)
  const int tid = threadIdx.x;
  const int w  = tid >> 5;
  const int ln = tid & 31;
  const int hf = ln >> 4;       // lane half (0: lanes0-15, 1: lanes16-31)
  const int mr = ln & 15;
  const int m0 = blockIdx.y * 128;
  const int n0 = blockIdx.x * 64;

  v8f acc[4];
  const v8f vzero = {0.f,0.f,0.f,0.f,0.f,0.f,0.f,0.f};
#pragma unroll
  for (int t = 0; t < 4; ++t) acc[t] = vzero;

  const int arow = tid >> 1;            // 0..127, 2 threads/row of A
  const int ap0  = (tid & 1) * 8;       // 8 bf16 pairs each
  const int bn   = tid >> 2;            // 0..63, 4 threads/col of B
  const int bp0  = (tid & 3) * 4;       // 4 bf16 pairs each
  const int ncol = n0 + bn;
  const int nclamp = (ncol < N) ? ncol : 0;     // valid address for edge cols
  const float nmask = (ncol < N) ? 1.f : 0.f;

  const float* __restrict__ Ar = A + (long)(m0 + arow) * K;

  auto mma_step = [&]() {
    // A 16x32 bf16 frag: v<4 -> pair hf*4+v ; v>=4 -> pair 8+hf*4+(v-4)
    FragAB a;
#pragma unroll
    for (int vv = 0; vv < 4; ++vv) a.u[vv]     = Asu[w * 16 + mr][hf * 4 + vv];
#pragma unroll
    for (int vv = 0; vv < 4; ++vv) a.u[4 + vv] = Asu[w * 16 + mr][8 + hf * 4 + vv];
#pragma unroll
    for (int nt = 0; nt < 4; ++nt) {
      // B 32x16 frag: N = mr, v<4 -> pair hf*4+v ; v>=4 -> pair 8+hf*4+(v-4)
      FragAB b;
#pragma unroll
      for (int vv = 0; vv < 8; ++vv) b.u[vv] = Bsu[nt * 16 + mr][(vv < 4) ? (hf * 4 + vv) : (4 + hf * 4 + vv)];
      acc[nt] = __builtin_amdgcn_wmma_f32_16x16x32_bf16(
          false, a.v, false, b.v, (short)0, acc[nt], false, false);
    }
  };

  const int KTf = K >> 5;               // full 32-wide K tiles
  for (int kt = 0; kt < KTf; ++kt) {
    const int kb = kt << 5;
    {   // A tile: straight-line loads, pack pairs, one ds_store_b32 per pair
#pragma unroll
      for (int i = 0; i < 8; ++i) {
        int k = kb + (ap0 + i) * 2;
        Asu[arow][ap0 + i] = f2bf_pack(Ar[k] * ascale, Ar[k + 1] * ascale);
      }
      if (kb + 32 < K) __builtin_prefetch(&Ar[kb + 32], 0, 0); // global_prefetch_b8
    }
    {   // B tile transposed into Bsu[n][kpair]; branch-free N edge
#pragma unroll
      for (int i = 0; i < 4; ++i) {
        int k = kb + (bp0 + i) * 2;
        float x0 = Bw[(long)k * N + nclamp] * nmask;
        float x1 = Bw[(long)(k + 1) * N + nclamp] * nmask;
        Bsu[bn][bp0 + i] = f2bf_pack(x0, x1);
      }
    }
    __syncthreads();
    mma_step();
    __syncthreads();
  }

  if (K & 31) {                         // tail tile: clamp addresses, select 0
    const int kb = KTf << 5;
#pragma unroll
    for (int i = 0; i < 8; ++i) {
      int k0 = kb + (ap0 + i) * 2, k1 = k0 + 1;
      float x0 = Ar[(k0 < K) ? k0 : (K - 1)];
      float x1 = Ar[(k1 < K) ? k1 : (K - 1)];
      x0 = (k0 < K) ? x0 * ascale : 0.f;
      x1 = (k1 < K) ? x1 * ascale : 0.f;
      Asu[arow][ap0 + i] = f2bf_pack(x0, x1);
    }
#pragma unroll
    for (int i = 0; i < 4; ++i) {
      int k0 = kb + (bp0 + i) * 2, k1 = k0 + 1;
      float x0 = Bw[(long)((k0 < K) ? k0 : (K - 1)) * N + nclamp];
      float x1 = Bw[(long)((k1 < K) ? k1 : (K - 1)) * N + nclamp];
      x0 = (k0 < K) ? x0 * nmask : 0.f;
      x1 = (k1 < K) ? x1 * nmask : 0.f;
      Bsu[bn][bp0 + i] = f2bf_pack(x0, x1);
    }
    __syncthreads();
    mma_step();
    __syncthreads();
  }

  // C/D 16x16 f32 layout: lane half hf -> rows hf*8+r, col = mr
#pragma unroll
  for (int nt = 0; nt < 4; ++nt) {
    int n = n0 + nt * 16 + mr;
    if (n >= N) continue;
    float bv = bias ? bias[n] : 0.f;
#pragma unroll
    for (int r = 0; r < 8; ++r) {
      int m = m0 + w * 16 + hf * 8 + r;
      C[(long)m * N + n] = epilogue<EPI>(acc[nt][r] + bv);
    }
  }
}

// --------------------------- attention (DH=4) ------------------------------
// One thread per (batch, head, query): online softmax over Lk keys.
__global__ void attn_kernel(const float* __restrict__ Q, const float* __restrict__ Km,
                            const float* __restrict__ Vm, float* __restrict__ O,
                            int Bn, int Lq, int Lk)
{
  int t = blockIdx.x * blockDim.x + threadIdx.x;
  if (t >= Bn * 4 * Lq) return;
  int qi = t % Lq, h = (t / Lq) & 3, b = t / (Lq * 4);
  const float* q = &Q[((long)b * Lq + qi) * 16 + h * 4];
  float q0 = q[0], q1 = q[1], q2 = q[2], q3 = q[3];
  float mx = -1e30f, sum = 0.f, a0 = 0, a1 = 0, a2 = 0, a3 = 0;
  for (int k = 0; k < Lk; ++k) {
    const float* kp = &Km[((long)b * Lk + k) * 16 + h * 4];
    const float* vp = &Vm[((long)b * Lk + k) * 16 + h * 4];
    float s = (q0 * kp[0] + q1 * kp[1] + q2 * kp[2] + q3 * kp[3]) * 0.5f; // 1/sqrt(4)
    if (s > mx) {
      float c = __expf(mx - s);
      sum *= c; a0 *= c; a1 *= c; a2 *= c; a3 *= c; mx = s;
    }
    float wgt = __expf(s - mx);
    sum += wgt; a0 += wgt * vp[0]; a1 += wgt * vp[1]; a2 += wgt * vp[2]; a3 += wgt * vp[3];
  }
  float inv = 1.f / sum;
  float* o = &O[((long)b * Lq + qi) * 16 + h * 4];
  o[0] = a0 * inv; o[1] = a1 * inv; o[2] = a2 * inv; o[3] = a3 * inv;
}

// --------------------- residual add + LayerNorm (1024) ---------------------
__global__ __launch_bounds__(256)
void add_ln_kernel(float* __restrict__ X, const float* __restrict__ D,
                   const float* __restrict__ g, const float* __restrict__ c)
{
  __shared__ float red[256];
  int row = blockIdx.x, tid = threadIdx.x;
  float v[4]; float s = 0.f;
#pragma unroll
  for (int i = 0; i < 4; ++i) {
    int col = tid + i * 256;
    v[i] = X[(long)row * 1024 + col] + D[(long)row * 1024 + col];
    s += v[i];
  }
  red[tid] = s; __syncthreads();
  for (int st = 128; st > 0; st >>= 1) { if (tid < st) red[tid] += red[tid + st]; __syncthreads(); }
  float mean = red[0] * (1.f / 1024.f);
  __syncthreads();
  s = 0.f;
#pragma unroll
  for (int i = 0; i < 4; ++i) { float d = v[i] - mean; s += d * d; }
  red[tid] = s; __syncthreads();
  for (int st = 128; st > 0; st >>= 1) { if (tid < st) red[tid] += red[tid + st]; __syncthreads(); }
  float rs = rsqrtf(red[0] * (1.f / 1024.f) + 1e-5f);
#pragma unroll
  for (int i = 0; i < 4; ++i) {
    int col = tid + i * 256;
    X[(long)row * 1024 + col] = (v[i] - mean) * rs * g[col] + c[col];
  }
}

// ------------------ 1024-pt radix-2 rFFT -> [abs, angle] -------------------
__global__ __launch_bounds__(512)
void fft_kernel(const float* __restrict__ Y, float* __restrict__ NF)
{
  __shared__ float re[1024], im[1024];
  int r = blockIdx.x, t = threadIdx.x;
  for (int i = t; i < 1024; i += 512) {
    unsigned rev = __brev((unsigned)i) >> 22;           // 10-bit bit reverse
    re[rev] = Y[(long)r * 1024 + i]; im[rev] = 0.f;
  }
  __syncthreads();
  for (int len = 2; len <= 1024; len <<= 1) {
    int half = len >> 1;
    int j = t % half, blk = t / half;
    int p = blk * len + j;
    float ang = -6.283185307179586f * (float)j / (float)len;
    float wr = __cosf(ang), wi = __sinf(ang);
    float ur = re[p], ui = im[p];
    float vr = re[p + half], vi = im[p + half];
    float tr = vr * wr - vi * wi, ti = vr * wi + vi * wr;
    re[p] = ur + tr; im[p] = ui + ti;
    re[p + half] = ur - tr; im[p + half] = ui - ti;
    __syncthreads();
  }
  for (int k = t; k <= 512; k += 512) {                 // 513 rfft bins
    float a = re[k], b = im[k];
    NF[(long)r * 1026 + k]       = sqrtf(a * a + b * b);
    NF[(long)r * 1026 + 513 + k] = atan2f(b, a);
  }
}

// ----------------------- graph gather * gate -> scatter --------------------
__global__ void scatter_kernel(const float* __restrict__ hv, const float* __restrict__ he,
                               const int* __restrict__ src, const int* __restrict__ dst,
                               float* __restrict__ agg, int e0, int ne)
{
  long t = (long)blockIdx.x * blockDim.x + threadIdx.x;
  if (t >= (long)ne * 256) return;
  int le = (int)(t >> 8), f = (int)(t & 255);
  int e = e0 + le;
  float m = hv[(long)src[e] * 256 + f] * he[(long)le * 256 + f];
  atomicAdd(&agg[(long)dst[e] * 256 + f], m);
}

// ------------------------------- misc --------------------------------------
__global__ void zero_kernel(unsigned* p, long n) {
  long t = (long)blockIdx.x * blockDim.x + threadIdx.x;
  if (t < n) p[t] = 0u;
}
__global__ void copy_kernel(float* __restrict__ d, const float* __restrict__ s, long n) {
  long t = (long)blockIdx.x * blockDim.x + threadIdx.x;
  if (t < n) d[t] = s[t];
}
__global__ void bcast_kernel(const float* __restrict__ S, float* __restrict__ Y) {
  int t = blockIdx.x * 256 + threadIdx.x;     // 16 x (8*1024)
  if (t < 16 * 8 * 1024) Y[t] = S[t % (8 * 1024)];
}
__global__ void colmax_kernel(const float* __restrict__ phi, unsigned* __restrict__ mx) {
  int t = blockIdx.x * 256 + threadIdx.x;
  if (t < 8192 * 8) atomicMax(&mx[t & 7], __float_as_uint(fabsf(phi[t])));
}
__global__ void phinorm_kernel(const float* __restrict__ phi, const unsigned* __restrict__ mx,
                               float* __restrict__ out) {
  int t = blockIdx.x * 256 + threadIdx.x;
  if (t < 8192 * 8) out[t] = phi[t] / __uint_as_float(mx[t & 7]);
}

// ------------------------------ host side ----------------------------------
static void gemm(hipStream_t s, const float* A, const float* B, const float* bias,
                 float* C, int M, int N, int K, int epi, float ascale = 1.0f)
{
  dim3 grid((N + 63) / 64, M / 128), blk(256);
  switch (epi) {
    case 0: gemm_wmma_bf16<0><<<grid, blk, 0, s>>>(A, B, bias, C, M, N, K, ascale); break;
    case 1: gemm_wmma_bf16<1><<<grid, blk, 0, s>>>(A, B, bias, C, M, N, K, ascale); break;
    case 2: gemm_wmma_bf16<2><<<grid, blk, 0, s>>>(A, B, bias, C, M, N, K, ascale); break;
    case 3: gemm_wmma_bf16<3><<<grid, blk, 0, s>>>(A, B, bias, C, M, N, K, ascale); break;
    case 4: gemm_wmma_bf16<4><<<grid, blk, 0, s>>>(A, B, bias, C, M, N, K, ascale); break;
  }
}

struct MhaP { const float *Wq,*Wk,*Wv,*Wo,*W1,*b1,*W2,*b2,*g1,*c1,*g2,*c2; };
static MhaP mha_at(void* const* din, int i) {
  MhaP p;
  p.Wq=(const float*)din[i+0];  p.Wk=(const float*)din[i+1];
  p.Wv=(const float*)din[i+2];  p.Wo=(const float*)din[i+3];
  p.W1=(const float*)din[i+4];  p.b1=(const float*)din[i+5];
  p.W2=(const float*)din[i+6];  p.b2=(const float*)din[i+7];
  p.g1=(const float*)din[i+8];  p.c1=(const float*)din[i+9];
  p.g2=(const float*)din[i+10]; p.c2=(const float*)din[i+11];
  return p;
}

// x:[rows_q,1024] updated in place; mem:[rows_kv,1024]
static void run_mha(hipStream_t s, float* x, int rows_q, int Lq,
                    const float* mem, int rows_kv, int Lk, const MhaP& p,
                    float* Q, float* K, float* V, float* AO, float* DELTA, float* FF1)
{
  int Bn = rows_q / Lq;
  gemm(s, x,   p.Wq, nullptr, Q, rows_q,  16, 1024, 0);
  gemm(s, mem, p.Wk, nullptr, K, rows_kv, 16, 1024, 0);
  gemm(s, mem, p.Wv, nullptr, V, rows_kv, 16, 1024, 0);
  int nthr = Bn * 4 * Lq;
  attn_kernel<<<(nthr + 255) / 256, 256, 0, s>>>(Q, K, V, AO, Bn, Lq, Lk);
  gemm(s, AO, p.Wo, nullptr, DELTA, rows_q, 1024, 16, 0);
  add_ln_kernel<<<rows_q, 256, 0, s>>>(x, DELTA, p.g1, p.c1);
  gemm(s, x,   p.W1, p.b1, FF1,   rows_q, 256, 1024, 1);
  gemm(s, FF1, p.W2, p.b2, DELTA, rows_q, 1024, 256, 0);
  add_ln_kernel<<<rows_q, 256, 0, s>>>(x, DELTA, p.g2, p.c2);
}

extern "C" void kernel_launch(void* const* d_in, const int* in_sizes, int n_in,
                              void* d_out, int out_size, void* d_ws, size_t ws_size,
                              hipStream_t stream)
{
  (void)in_sizes; (void)n_in; (void)out_size; (void)ws_size;
  const float* accY = (const float*)d_in[0];   // 8192 x 1024
  const float* cpsd = (const float*)d_in[1];   // 131072 x 513
  const int*   srcI = (const int*)d_in[2];
  const int*   dstI = (const int*)d_in[3];
  MhaP enc0 = mha_at(d_in, 4), enc1 = mha_at(d_in, 16);
  const float* Sseed = (const float*)d_in[28];
  const float* rW1 = (const float*)d_in[29]; const float* rb1 = (const float*)d_in[30];
  const float* rW2 = (const float*)d_in[31]; const float* rb2 = (const float*)d_in[32];
  MhaP pma  = mha_at(d_in, 33);
  MhaP dec0 = mha_at(d_in, 45), dec1 = mha_at(d_in, 57);
  const float *Wn1=(const float*)d_in[69], *bn1=(const float*)d_in[70],
              *We11=(const float*)d_in[71], *be11=(const float*)d_in[72],
              *We21=(const float*)d_in[73], *be21=(const float*)d_in[74],
              *Wo1=(const float*)d_in[75], *bo1=(const float*)d_in[76];
  const float *Wn2=(const float*)d_in[77], *bn2=(const float*)d_in[78],
              *We12=(const float*)d_in[79], *be12=(const float*)d_in[80],
              *We22=(const float*)d_in[81], *be22=(const float*)d_in[82],
              *Wo2=(const float*)d_in[83], *bo2=(const float*)d_in[84];
  const float *mW1=(const float*)d_in[85], *mb1=(const float*)d_in[86],
              *mW2=(const float*)d_in[87], *mb2=(const float*)d_in[88],
              *mW3=(const float*)d_in[89], *mb3=(const float*)d_in[90];

  // workspace carve-out (~246 MB of f32)
  float* ws = (float*)d_ws; size_t off = 0;
  auto alloc = [&](size_t n) { float* p = ws + off; off += n; return p; };
  float* XA    = alloc(8192L * 1024);
  float* DELTA = alloc(8192L * 1024);
  float* MEMB  = alloc(8192L * 1024);
  float* FF1   = alloc(8192L * 256);
  float* Qb    = alloc(8192L * 16);
  float* Kb    = alloc(8192L * 16);
  float* Vb    = alloc(8192L * 16);
  float* AO    = alloc(8192L * 16);
  float* YA    = alloc(128L * 1024);
  float* NF    = alloc(8192L * 1026);
  float* HV    = alloc(8192L * 256);
  float* AGG   = alloc(8192L * 256);
  float* Hb    = alloc(8192L * 256);
  float* Zb    = alloc(8192L * 256);
  float* E1    = alloc(32768L * 256);
  float* HE    = alloc(32768L * 256);
  float* PHI   = alloc(8192L * 8);
  unsigned* MX = (unsigned*)alloc(8);

  // ---- SetTransformer encoder (2 SAB layers, self-attn) ----
  copy_kernel<<<(8192L * 1024 + 255) / 256, 256, 0, stream>>>(XA, accY, 8192L * 1024);
  run_mha(stream, XA, 8192, 512, XA, 8192, 512, enc0, Qb, Kb, Vb, AO, DELTA, FF1);
  run_mha(stream, XA, 8192, 512, XA, 8192, 512, enc1, Qb, Kb, Vb, AO, DELTA, FF1);

  // ---- decoder: mem = relu(x@rW1+rb1)@rW2+rb2 ; PMA(seeds) ; 2 SAB ----
  gemm(stream, XA,  rW1, rb1, FF1,  8192, 256, 1024, 1);
  gemm(stream, FF1, rW2, rb2, MEMB, 8192, 1024, 256, 0);
  bcast_kernel<<<(16 * 8 * 1024 + 255) / 256, 256, 0, stream>>>(Sseed, YA);
  run_mha(stream, YA, 128, 8, MEMB, 8192, 512, pma,  Qb, Kb, Vb, AO, DELTA, FF1);
  run_mha(stream, YA, 128, 8, YA,   128,  8,   dec0, Qb, Kb, Vb, AO, DELTA, FF1);
  run_mha(stream, YA, 128, 8, YA,   128,  8,   dec1, Qb, Kb, Vb, AO, DELTA, FF1);
  // q = reshape(y) : identical flat layout -> direct copy to d_out[0:131072]
  copy_kernel<<<(131072 + 255) / 256, 256, 0, stream>>>((float*)d_out, YA, 131072);

  // ---- node FFT features: [abs, angle] of 1024-pt rfft ----
  fft_kernel<<<8192, 512, 0, stream>>>(accY, NF);

  const float invpi = 1.0f / 3.14f;    // edge_in = cpsd_angle / 3.14 folded into A-scale
  // ---- CFConv 1 ----
  gemm(stream, NF, Wn1, bn1, HV, 8192, 256, 1026, 0);
  zero_kernel<<<(8192L * 256 + 255) / 256, 256, 0, stream>>>((unsigned*)AGG, 8192L * 256);
  for (int c = 0; c < 4; ++c) {
    int e0 = c * 32768;
    gemm(stream, cpsd + (long)e0 * 513, We11, be11, E1, 32768, 256, 513, 2, invpi);
    gemm(stream, E1, We21, be21, HE, 32768, 256, 256, 2);
    scatter_kernel<<<32768, 256, 0, stream>>>(HV, HE, srcI, dstI, AGG, e0, 32768);
  }
  gemm(stream, AGG, Wo1, bo1, Hb, 8192, 256, 256, 3);   // tanh(ssp(.))

  // ---- CFConv 2 ----
  gemm(stream, Hb, Wn2, bn2, HV, 8192, 256, 256, 0);
  zero_kernel<<<(8192L * 256 + 255) / 256, 256, 0, stream>>>((unsigned*)AGG, 8192L * 256);
  for (int c = 0; c < 4; ++c) {
    int e0 = c * 32768;
    gemm(stream, cpsd + (long)e0 * 513, We12, be12, E1, 32768, 256, 513, 2, invpi);
    gemm(stream, E1, We22, be22, HE, 32768, 256, 256, 2);
    scatter_kernel<<<32768, 256, 0, stream>>>(HV, HE, srcI, dstI, AGG, e0, 32768);
  }
  gemm(stream, AGG, Wo2, bo2, Zb, 8192, 256, 256, 3);

  // ---- MLP decoder: tanh x3 then linear head ----
  gemm(stream, Zb, mW1, mb1, Hb, 8192, 256, 256, 4);
  gemm(stream, Hb, mW2, mb2, Zb, 8192, 256, 256, 4);
  gemm(stream, Zb, mW2, mb2, Hb, 8192, 256, 256, 4);
  gemm(stream, Hb, mW3, mb3, PHI, 8192, 8, 256, 0);

  // ---- phi / max(|phi|, axis=0) -> d_out[131072:] ----
  zero_kernel<<<1, 256, 0, stream>>>(MX, 8);
  colmax_kernel<<<(8192 * 8 + 255) / 256, 256, 0, stream>>>(PHI, MX);
  phinorm_kernel<<<(8192 * 8 + 255) / 256, 256, 0, stream>>>(PHI, MX, (float*)d_out + 131072);
}
